// ArcticMoE_1864015807089
// MI455X (gfx1250) — compile-verified, matrix operand order checked by reading, not run on hardware
//
#include <hip/hip_runtime.h>
#include <hip/hip_bf16.h>

// ---------------------------------------------------------------------------
// MoE (Arctic-style) for gfx1250: bf16 WMMA grouped GEMMs with async
// global->LDS double-buffered staging + deterministic counting-sort routing.
// fp32 in / fp32 out; weights & activations converted to bf16 once per launch.
// ---------------------------------------------------------------------------

#define NUM_T   4096       // tokens (2*2048)
#define MODEL_D 2048
#define INTER_D 1024
#define NEXP    8
#define PAIRS   8192       // NUM_T * TOP_K
#define RPAD    9216       // PAIRS + NEXP*128 worst-case padding
#define TILE    128
#define KSTAGE  64         // K elements staged per async stage (2 WMMA steps)

typedef __attribute__((ext_vector_type(16))) __bf16 v16bf;
typedef __attribute__((ext_vector_type(8)))  float  v8f;

union FragBF { uint4 q[2]; v16bf v; };

__device__ __forceinline__ unsigned short f2bf(float f) {
    union { float f; unsigned u; } x; x.f = f;
    unsigned r = x.u + 0x7FFFu + ((x.u >> 16) & 1u);   // round-nearest-even
    return (unsigned short)(r >> 16);
}
__device__ __forceinline__ unsigned pk2(float lo, float hi) {
    return (unsigned)f2bf(lo) | ((unsigned)f2bf(hi) << 16);
}

// ---- CDNA5 async global->LDS copy (ASYNCcnt-tracked, no VGPR round trip).
// INST_OFFSET is added to BOTH the global address and the LDS address, so the
// four 16B chunks per matrix share one base address pair (no per-chunk v_add).
template <int OFF>
__device__ __forceinline__ void async_copy_b128(unsigned lds_off, const void* gptr) {
    asm volatile("global_load_async_to_lds_b128 %0, %1, off offset:%c2"
                 :: "v"(lds_off), "v"((unsigned long long)(size_t)gptr), "i"(OFF)
                 : "memory");
}
__device__ __forceinline__ void wait_async_le8() {
    asm volatile("s_wait_asynccnt 0x8" ::: "memory");
}
__device__ __forceinline__ void wait_async_0() {
    asm volatile("s_wait_asynccnt 0x0" ::: "memory");
}
__device__ __forceinline__ unsigned lds_off32(const void* p) {
    return (unsigned)(size_t)p;   // generic ptr low 32 bits = wave-relative LDS addr
}

// ---------------------------------------------------------------------------
// 1) Router: logits -> softmax -> top-2 -> normalized scores. One wave/token.
// ---------------------------------------------------------------------------
__launch_bounds__(256)
__global__ void router_topk(const float* __restrict__ hs,
                            const float* __restrict__ rg,
                            int* __restrict__ pexp, float* __restrict__ pscore)
{
    const int lane = threadIdx.x & 31;
    const int wave = threadIdx.x >> 5;
    const int t    = blockIdx.x * 8 + wave;

    float acc[NEXP];
#pragma unroll
    for (int e = 0; e < NEXP; ++e) acc[e] = 0.f;

    const float* x = hs + (size_t)t * MODEL_D;
    for (int k = lane; k < MODEL_D; k += 32) {
        float xv = x[k];
#pragma unroll
        for (int e = 0; e < NEXP; ++e) acc[e] += xv * rg[e * MODEL_D + k];
    }
#pragma unroll
    for (int e = 0; e < NEXP; ++e)
        for (int off = 16; off > 0; off >>= 1)
            acc[e] += __shfl_down(acc[e], off, 32);

    if (lane == 0) {
        float m = acc[0];
#pragma unroll
        for (int e = 1; e < NEXP; ++e) m = fmaxf(m, acc[e]);
        float p[NEXP];
#pragma unroll
        for (int e = 0; e < NEXP; ++e) p[e] = __expf(acc[e] - m);
        int i0 = 0;
#pragma unroll
        for (int e = 1; e < NEXP; ++e) if (p[e] > p[i0]) i0 = e;
        int i1 = (i0 == 0) ? 1 : 0;
#pragma unroll
        for (int e = 0; e < NEXP; ++e) if (e != i0 && p[e] > p[i1]) i1 = e;
        float s = p[i0] + p[i1];
        pexp[t]          = i0;
        pexp[NUM_T + t]  = i1;
        pscore[t]         = p[i0] / s;
        pscore[NUM_T + t] = p[i1] / s;
    }
}

// ---------------------------------------------------------------------------
// 2) Stable per-expert rank (counting sort order == jax stable argsort).
// ---------------------------------------------------------------------------
__launch_bounds__(256)
__global__ void rank_count(const int* __restrict__ pexp,
                           int* __restrict__ prank, int* __restrict__ counts)
{
    __shared__ int wsum[8];
    __shared__ int running;
    const int e = blockIdx.x, tid = threadIdx.x;
    const int lane = tid & 31, w = tid >> 5;
    if (tid == 0) running = 0;
    __syncthreads();
    for (int base = 0; base < PAIRS; base += 256) {
        const int p = base + tid;
        const bool m = (pexp[p] == e);
        unsigned bal = (unsigned)__ballot(m);             // wave32: low 32 bits
        if (lane == 0) wsum[w] = __popc(bal);
        __syncthreads();
        int off = running;
        for (int i = 0; i < w; ++i) off += wsum[i];
        const int before = __popc(bal & ((1u << lane) - 1u));
        if (m) prank[p] = off + before;
        __syncthreads();
        if (tid == 0) {
            int tot = 0;
            for (int i = 0; i < 8; ++i) tot += wsum[i];
            running += tot;
        }
        __syncthreads();
    }
    if (tid == 0) counts[e] = running;
}

__global__ void scan_pad(const int* __restrict__ counts, int* __restrict__ padOff)
{
    if (threadIdx.x == 0 && blockIdx.x == 0) {
        int o = 0;
        for (int e = 0; e < NEXP; ++e) {
            padOff[e] = o;
            o += ((counts[e] + TILE - 1) / TILE) * TILE;
        }
        padOff[NEXP] = o;
    }
}

__launch_bounds__(256)
__global__ void scatter_rows(const int* __restrict__ pexp, const int* __restrict__ prank,
                             const int* __restrict__ padOff,
                             int* __restrict__ prow, int* __restrict__ rowtok)
{
    const int p = blockIdx.x * 256 + threadIdx.x;   // 0..8191
    const int e = pexp[p];
    const int r = padOff[e] + prank[p];
    prow[p]   = r;
    rowtok[r] = p & (NUM_T - 1);
}

// ---------------------------------------------------------------------------
// 3) Gather token rows in sorted order, fp32 -> packed bf16; zero pad rows.
// ---------------------------------------------------------------------------
__launch_bounds__(256)
__global__ void gather_cvt(const float* __restrict__ hs,
                           const int* __restrict__ rowtok,
                           const int* __restrict__ padOff,
                           const int* __restrict__ counts,
                           unsigned short* __restrict__ xbf)
{
    const int r = blockIdx.x;
    if (r >= padOff[NEXP]) return;
    int e = 0;
#pragma unroll
    for (int i = 0; i < NEXP; ++i) if (r >= padOff[i]) e = i;
    const bool real = (r - padOff[e]) < counts[e];
    unsigned short* drow = xbf + (size_t)r * MODEL_D;
    const int c = threadIdx.x * 8;                   // 256 threads * 8 = 2048
    if (real) {
        const int t = rowtok[r];
        const float4* srow = (const float4*)(hs + (size_t)t * MODEL_D);
        float4 a = srow[(c >> 2) + 0];
        float4 b = srow[(c >> 2) + 1];
        uint4 v;
        v.x = pk2(a.x, a.y); v.y = pk2(a.z, a.w);
        v.z = pk2(b.x, b.y); v.w = pk2(b.z, b.w);
        *(uint4*)(drow + c) = v;
    } else {
        uint4 z; z.x = z.y = z.z = z.w = 0u;
        *(uint4*)(drow + c) = z;
    }
}

// ---------------------------------------------------------------------------
// 4) Weight convert + transpose: fp32 [E][K][Nc] -> bf16 [E][J][K].
//    inter=1 de-interleaves gate/up: j<Nc/2 -> col 2j (gate), else 2(j-Nc/2)+1.
// ---------------------------------------------------------------------------
__launch_bounds__(256)
__global__ void transpose_cvt(const float* __restrict__ src,
                              unsigned short* __restrict__ dst,
                              int K, int Nc, int inter)
{
    __shared__ unsigned short tile[32][33];
    const int e  = blockIdx.z;
    const int k0 = blockIdx.x * 32;
    const int j0 = blockIdx.y * 32;
    const float* s = src + (size_t)e * K * Nc;
    unsigned short* d = dst + (size_t)e * Nc * K;

#pragma unroll
    for (int i = 0; i < 4; ++i) {                    // read 32x32 fp32, cvt
        const int idx = threadIdx.x + i * 256;
        const int kk = idx >> 5, jj = idx & 31;
        const int j = j0 + jj;
        const int col = inter ? ((j < (Nc >> 1)) ? 2 * j : 2 * (j - (Nc >> 1)) + 1) : j;
        tile[kk][jj] = f2bf(s[(size_t)(k0 + kk) * Nc + col]);
    }
    __syncthreads();
#pragma unroll
    for (int i = 0; i < 2; ++i) {                    // write transposed, 2 bf16/store
        const int idx = threadIdx.x + i * 256;       // 512 dword writes
        const int jj = idx >> 4;
        const int kp = (idx & 15) * 2;
        unsigned v = (unsigned)tile[kp][jj] | ((unsigned)tile[kp + 1][jj] << 16);
        *(unsigned*)&d[(size_t)(j0 + jj) * K + k0 + kp] = v;
    }
}

// ---------------------------------------------------------------------------
// 5) Grouped GEMM: C[128x128] per block, 8 waves (2M x 4N), bf16 WMMA.
//    Double-buffered async global->LDS staging, KSTAGE=64 per stage.
//    A: [RPAD][KDIM] bf16 (rows grouped+padded by expert)
//    B: [E][2048][KDIM] bf16 (pre-transposed, K contiguous)
//    MODE 1: split columns into gate[.,0:1024] / up[.,0:1024]
//    MODE 2: dense out_rows [RPAD][2048]
// ---------------------------------------------------------------------------
template <int MODE, int KDIM>
__launch_bounds__(256)
__global__ void moe_gemm(const unsigned short* __restrict__ A,
                         const unsigned short* __restrict__ B,
                         float* __restrict__ out0, float* __restrict__ out1,
                         const int* __restrict__ padOff)
{
    constexpr int NST = KDIM / KSTAGE;
    __shared__ unsigned short As[2][TILE * KSTAGE];   // 16KB per buffer
    __shared__ unsigned short Bs[2][TILE * KSTAGE];

    const int tcol  = blockIdx.x;
    const int trow  = blockIdx.y;
    const int rbase = trow * TILE;
    if (rbase >= padOff[NEXP]) return;               // beyond active rows
    int e = 0;
#pragma unroll
    for (int i = 0; i < NEXP; ++i) if (rbase >= padOff[i]) e = i;

    const int tid  = threadIdx.x;
    const int lane = tid & 31, wave = tid >> 5;
    const int wm = wave >> 2, wn = wave & 3;         // 2 waves in M, 4 in N
    const int g  = lane >> 4, l16 = lane & 15;
    const int klo = 8 * g, khi = 16 + 8 * g;         // per-lane K chunks (bf16)

    // staging split: thread -> row (tid>>1), K half ((tid&1)*32) of the 64-slab
    const int srow  = tid >> 1;
    const int skoff = (tid & 1) * 32;
    const unsigned short* Ag = A + (size_t)(rbase + srow) * KDIM + skoff;
    const unsigned short* Bg = B + ((size_t)e * 2048 + tcol * TILE + srow) * KDIM + skoff;
    unsigned lA[2], lB[2];
#pragma unroll
    for (int b = 0; b < 2; ++b) {
        lA[b] = lds_off32(&As[b][srow * KSTAGE + skoff]);
        lB[b] = lds_off32(&Bs[b][srow * KSTAGE + skoff]);
    }

    v8f acc[4][2];
#pragma unroll
    for (int fm = 0; fm < 4; ++fm)
#pragma unroll
        for (int fn = 0; fn < 2; ++fn)
#pragma unroll
            for (int i = 0; i < 8; ++i) acc[fm][fn][i] = 0.f;

    auto issue = [&](int buf, int stage) {
        const unsigned short* ga = Ag + (size_t)stage * KSTAGE;
        const unsigned short* gb = Bg + (size_t)stage * KSTAGE;
        async_copy_b128< 0>(lA[buf], ga);            // offset applies to LDS and
        async_copy_b128<16>(lA[buf], ga);            // global alike: one base,
        async_copy_b128<32>(lA[buf], ga);            // four immediate offsets
        async_copy_b128<48>(lA[buf], ga);
        async_copy_b128< 0>(lB[buf], gb);
        async_copy_b128<16>(lB[buf], gb);
        async_copy_b128<32>(lB[buf], gb);
        async_copy_b128<48>(lB[buf], gb);
    };

    auto compute = [&](int buf) {
        const unsigned short* Ab = As[buf];
        const unsigned short* Bb = Bs[buf];
#pragma unroll
        for (int ks = 0; ks < 2; ++ks) {              // two 16x16x32 K-steps
            const int kb = ks * 32;
            FragBF af[4], bf[2];
#pragma unroll
            for (int fm = 0; fm < 4; ++fm) {
                const unsigned short* p = Ab + (wm * 64 + fm * 16 + l16) * KSTAGE + kb;
                af[fm].q[0] = *(const uint4*)(p + klo);
                af[fm].q[1] = *(const uint4*)(p + khi);
            }
#pragma unroll
            for (int fn = 0; fn < 2; ++fn) {
                const unsigned short* p = Bb + (wn * 32 + fn * 16 + l16) * KSTAGE + kb;
                bf[fn].q[0] = *(const uint4*)(p + klo);
                bf[fn].q[1] = *(const uint4*)(p + khi);
            }
#pragma unroll
            for (int fm = 0; fm < 4; ++fm)
#pragma unroll
                for (int fn = 0; fn < 2; ++fn)
                    acc[fm][fn] = __builtin_amdgcn_wmma_f32_16x16x32_bf16(
                        false, af[fm].v, false, bf[fn].v,
                        (short)0, acc[fm][fn], false, false);
        }
    };

    issue(0, 0);                                      // prologue
#pragma unroll 2
    for (int s = 0; s < NST - 1; ++s) {               // steady state: no branches
        issue((s + 1) & 1, s + 1);
        wait_async_le8();                             // in-order: oldest 8 done
        __syncthreads();
        compute(s & 1);
        __syncthreads();                              // buffer reusable next iter
    }
    wait_async_0();                                   // peeled final stage
    __syncthreads();
    compute((NST - 1) & 1);

    // Epilogue: C layout = VGPR i -> M=i (lanes 0-15) / M=8+i (lanes 16-31)
#pragma unroll
    for (int fm = 0; fm < 4; ++fm) {
        const int mrow = rbase + wm * 64 + fm * 16 + (g ? 8 : 0);
#pragma unroll
        for (int fn = 0; fn < 2; ++fn) {
            const int j = tcol * TILE + wn * 32 + fn * 16 + l16;
#pragma unroll
            for (int i = 0; i < 8; ++i) {
                const float val = acc[fm][fn][i];
                const int r = mrow + i;
                if (MODE == 1) {
                    if (j < INTER_D) out0[(size_t)r * INTER_D + j] = val;
                    else             out1[(size_t)r * INTER_D + (j - INTER_D)] = val;
                } else {
                    out0[(size_t)r * MODEL_D + j] = val;
                }
            }
        }
    }
}

// ---------------------------------------------------------------------------
// 6) h = up * silu(gate)  (GATE_SCALE=1, UP_SCALE=0), packed bf16 output.
// ---------------------------------------------------------------------------
__launch_bounds__(256)
__global__ void act_silu(const float* __restrict__ gate,
                         const float* __restrict__ up,
                         unsigned short* __restrict__ h)
{
    const size_t i = ((size_t)blockIdx.x * 256 + threadIdx.x) * 2;
    const float g0 = gate[i], g1 = gate[i + 1];
    const float u0 = up[i],   u1 = up[i + 1];
    const float h0 = u0 * (g0 / (1.f + __expf(-g0)));
    const float h1 = u1 * (g1 / (1.f + __expf(-g1)));
    *(unsigned*)(h + i) = pk2(h0, h1);
}

// ---------------------------------------------------------------------------
// 7) Combine: out[t] = s0*out_rows[r0] + s1*out_rows[r1] (no atomics).
// ---------------------------------------------------------------------------
__launch_bounds__(256)
__global__ void combine(const float* __restrict__ orows,
                        const int* __restrict__ prow,
                        const float* __restrict__ pscore,
                        float* __restrict__ out)
{
    const int t = blockIdx.x;
    const int r0 = prow[t], r1 = prow[NUM_T + t];
    const float s0 = pscore[t], s1 = pscore[NUM_T + t];
    const float4* a = (const float4*)(orows + (size_t)r0 * MODEL_D);
    const float4* b = (const float4*)(orows + (size_t)r1 * MODEL_D);
    float4* o = (float4*)(out + (size_t)t * MODEL_D);
    for (int c = threadIdx.x; c < MODEL_D / 4; c += 256) {
        float4 x = a[c], y = b[c], z;
        z.x = s0 * x.x + s1 * y.x;
        z.y = s0 * x.y + s1 * y.y;
        z.z = s0 * x.z + s1 * y.z;
        z.w = s0 * x.w + s1 * y.w;
        o[c] = z;
    }
}

// ---------------------------------------------------------------------------
extern "C" void kernel_launch(void* const* d_in, const int* in_sizes, int n_in,
                              void* d_out, int out_size, void* d_ws, size_t ws_size,
                              hipStream_t stream)
{
    const float* hs  = (const float*)d_in[0];   // [2,2048,2048] fp32
    const float* rg  = (const float*)d_in[1];   // [8,2048] fp32
    const float* wgu = (const float*)d_in[2];   // [8,2048,2048] fp32 (interleaved gate/up)
    const float* wdn = (const float*)d_in[3];   // [8,1024,2048] fp32
    float* out = (float*)d_out;
    char* ws = (char*)d_ws;

    // ---- workspace layout (~233 MB) ----
    int*   counts = (int*)(ws + 0);
    int*   padOff = (int*)(ws + 64);
    int*   pexp   = (int*)(ws + 1024);
    float* pscore = (float*)(ws + 1024 + 1 * 32768);
    int*   prank  = (int*)(ws + 1024 + 2 * 32768);
    int*   prow   = (int*)(ws + 1024 + 3 * 32768);
    int*   rowtok = (int*)(ws + 1024 + 4 * 32768);
    size_t o = 262144;
    unsigned short* xbf = (unsigned short*)(ws + o); o += (size_t)RPAD * MODEL_D * 2;
    unsigned short* Wt  = (unsigned short*)(ws + o); o += (size_t)NEXP * 2048 * 2048 * 2;
    unsigned short* Dt  = (unsigned short*)(ws + o); o += (size_t)NEXP * 2048 * 1024 * 2;
    const size_t o_gate = o;
    float* gate = (float*)(ws + o); o += (size_t)RPAD * INTER_D * 4;
    float* up   = (float*)(ws + o); o += (size_t)RPAD * INTER_D * 4;
    unsigned short* hbf = (unsigned short*)(ws + o); o += (size_t)RPAD * INTER_D * 2;
    float* orows = (float*)(ws + o_gate);           // aliases gate+up (dead by then)

    // ---- routing ----
    router_topk<<<NUM_T / 8, 256, 0, stream>>>(hs, rg, pexp, pscore);
    rank_count<<<NEXP, 256, 0, stream>>>(pexp, prank, counts);
    scan_pad<<<1, 32, 0, stream>>>(counts, padOff);
    scatter_rows<<<PAIRS / 256, 256, 0, stream>>>(pexp, prank, padOff, prow, rowtok);
    gather_cvt<<<RPAD, 256, 0, stream>>>(hs, rowtok, padOff, counts, xbf);

    // ---- weight convert/transpose to bf16 [E][N][K] ----
    transpose_cvt<<<dim3(MODEL_D / 32, 2048 / 32, NEXP), 256, 0, stream>>>(
        wgu, Wt, MODEL_D, 2048, 1);
    transpose_cvt<<<dim3(INTER_D / 32, MODEL_D / 32, NEXP), 256, 0, stream>>>(
        wdn, Dt, INTER_D, MODEL_D, 0);

    // ---- grouped GEMM 1: x @ W_gate_up -> gate, up ----
    moe_gemm<1, MODEL_D><<<dim3(2048 / TILE, RPAD / TILE), 256, 0, stream>>>(
        xbf, Wt, gate, up, padOff);

    // ---- activation ----
    act_silu<<<(RPAD * INTER_D) / 512, 256, 0, stream>>>(gate, up, hbf);

    // ---- grouped GEMM 2: h @ W_down -> out_rows ----
    moe_gemm<2, INTER_D><<<dim3(MODEL_D / TILE, RPAD / TILE), 256, 0, stream>>>(
        hbf, Dt, orows, nullptr, padOff);

    // ---- weighted combine back to tokens ----
    combine<<<NUM_T, 256, 0, stream>>>(orows, prow, pscore, out);

    (void)in_sizes; (void)n_in; (void)out_size; (void)ws_size;
}
